// SamSdpaVisionAttentionSplit_45329084842403
// MI455X (gfx1250) — matrix operand order repaired and checked by reading, not compiled, not run on hardware
//
#include <hip/hip_runtime.h>
#include <hip/hip_bf16.h>

// ---------------------------------------------------------------------------
// SAM vision attention (decomposed rel-pos) for gfx1250 / MI455X.
// B=1, H=W=64, S=4096, C=768, HEADS=12, HD=64.
// All matmuls run on v_wmma_f32_16x16x32_f16 (f16 in, f32 accumulate),
// with 2x2 register-tile blocking so each fragment load feeds 2 WMMAs.
// ---------------------------------------------------------------------------

typedef __attribute__((ext_vector_type(16))) _Float16 v16h;
typedef __attribute__((ext_vector_type(8)))  _Float16 v8h;
typedef __attribute__((ext_vector_type(8)))  float    v8f;

#define S_LEN   4096
#define CDIM    768
#define HEADS   12
#define HD      64
#define HEAD_E  262144   // 4096*64 elements per head

// --- WMMA fragment loaders (ISA 7.12.2 layouts, wave32) --------------------
// A (16x32 f16): lane L holds row M=L%16; half-wave grp=L/16 selects the
// K-pairs {grp*8..grp*8+7} and {16+grp*8..16+grp*8+7} -> two b128 loads.
__device__ __forceinline__ v16h load_frag_a(const _Float16* __restrict__ base,
                                            int ld, int mrow, int grp, int k0) {
  const _Float16* p = base + (size_t)mrow * ld + k0 + grp * 8;
  union { v16h v; v8h h[2]; } u;
  u.h[0] = *(const v8h*)(p);
  u.h[1] = *(const v8h*)(p + 16);
  return u.v;
}
// B (32x16 f16) from B^T row-major [N][K]: lane L holds col N=L%16; grp
// selects K {grp*8..grp*8+7} and {16+grp*8..16+grp*8+7} -> two b128 loads.
__device__ __forceinline__ v16h load_frag_b(const _Float16* __restrict__ base,
                                            int ld, int ncol, int grp, int k0) {
  const _Float16* p = base + (size_t)ncol * ld + k0 + grp * 8;
  union { v16h v; v8h h[2]; } u;
  u.h[0] = *(const v8h*)(p);
  u.h[1] = *(const v8h*)(p + 16);
  return u.v;
}

#define WMMA_F16(a, b, c) \
  __builtin_amdgcn_wmma_f32_16x16x32_f16(false, (a), false, (b), (short)0, (c), false, false)

// --- elementwise converters -------------------------------------------------
__global__ void k_f32_to_f16(const float* __restrict__ in, _Float16* __restrict__ out, int n) {
  int i = blockIdx.x * 256 + threadIdx.x;
  if (i < n) out[i] = (_Float16)in[i];
}

// Rh/Rw gather: out[q][k][d] = rel[(q - k + 63)][d], 64*64*64 elements, -> f16
__global__ void k_gather_rel(const float* __restrict__ rel, _Float16* __restrict__ out) {
  int i = blockIdx.x * 256 + threadIdx.x;      // 262144 total
  int d  = i & 63;
  int kk = (i >> 6) & 63;
  int qq = i >> 12;
  out[i] = (_Float16)rel[(size_t)(qq - kk + 63) * 64 + d];
}

// --- QKV projection: y = X * W^T + b, scattered into per-head layouts ------
// 2x2 register tiling: one wave -> 32x32 output block.
// mode 0: q[head][s][d] f16     mode 1: k[head][s][d] f16 scaled by 1/8
// mode 2: vT[head][d][s] f16 (transposed for the PV GEMM)
__global__ __launch_bounds__(128) void k_qkv_gemm(
    const _Float16* __restrict__ X, const _Float16* __restrict__ W,
    const float* __restrict__ bias, _Float16* __restrict__ dst, int mode) {
  int wid  = threadIdx.x >> 5, lane = threadIdx.x & 31;
  int g    = blockIdx.x * 4 + wid;             // 3072 waves: 128 Mg x 24 Ng
  int m0   = (g / 24) * 32, n0 = (g % 24) * 32;
  int mrow = lane & 15, grp = (lane >> 4) & 1, ncol = lane & 15;

  const _Float16* A0 = X + (size_t)m0 * CDIM;
  const _Float16* A1 = X + (size_t)(m0 + 16) * CDIM;
  const _Float16* B0 = W + (size_t)n0 * CDIM;
  const _Float16* B1 = W + (size_t)(n0 + 16) * CDIM;
  v8f c[2][2] = {};
#pragma unroll 2
  for (int k0 = 0; k0 < CDIM; k0 += 32) {
    v16h a0 = load_frag_a(A0, CDIM, mrow, grp, k0);
    v16h a1 = load_frag_a(A1, CDIM, mrow, grp, k0);
    v16h b0 = load_frag_b(B0, CDIM, ncol, grp, k0);
    v16h b1 = load_frag_b(B1, CDIM, ncol, grp, k0);
    c[0][0] = WMMA_F16(a0, b0, c[0][0]);
    c[0][1] = WMMA_F16(a0, b1, c[0][1]);
    c[1][0] = WMMA_F16(a1, b0, c[1][0]);
    c[1][1] = WMMA_F16(a1, b1, c[1][1]);
  }
#pragma unroll
  for (int nt = 0; nt < 2; nt++) {
    int n = n0 + nt * 16 + ncol;
    float bv = bias[n];
    int head = n >> 6, d = n & 63;
#pragma unroll
    for (int mt = 0; mt < 2; mt++) {
#pragma unroll
      for (int i = 0; i < 8; i++) {
        int s = m0 + mt * 16 + i + 8 * grp;
        float y = c[mt][nt][i] + bv;
        if (mode == 0)      dst[(size_t)head * HEAD_E + (size_t)s * HD + d] = (_Float16)y;
        else if (mode == 1) dst[(size_t)head * HEAD_E + (size_t)s * HD + d] = (_Float16)(y * 0.125f);
        else                dst[(size_t)head * HEAD_E + (size_t)d * S_LEN + s] = (_Float16)y;
      }
    }
  }
}

// --- rel_h[h][q=qh*64+qw][kh] = sum_d q[h][qh*64+qw][d] * Rh[qh][kh][d] -----
// Per (h,qh): 64(qw) x 64(kh) product; one wave does a 32x32 sub-block.
__global__ __launch_bounds__(128) void k_relh_gemm(
    const _Float16* __restrict__ Q, const _Float16* __restrict__ Rh,
    float* __restrict__ out) {
  int wid = threadIdx.x >> 5, lane = threadIdx.x & 31;
  int g = blockIdx.x * 4 + wid;                // 3072 = 12h * 64qh * 4 sub
  int h = g / 256, rem = g % 256;
  int qh = rem / 4, sub = rem % 4;
  int qwg = sub >> 1, khg = sub & 1;
  int mrow = lane & 15, grp = (lane >> 4) & 1, ncol = lane & 15;

  const _Float16* Ab = Q + (size_t)h * HEAD_E + (size_t)qh * 4096 + (size_t)qwg * 32 * HD;
  const _Float16* Bb = Rh + (size_t)qh * 4096 + (size_t)khg * 32 * HD;
  v8f c[2][2] = {};
#pragma unroll
  for (int k0 = 0; k0 < HD; k0 += 32) {
    v16h a0 = load_frag_a(Ab,            HD, mrow, grp, k0);
    v16h a1 = load_frag_a(Ab + 16 * HD,  HD, mrow, grp, k0);
    v16h b0 = load_frag_b(Bb,            HD, ncol, grp, k0);
    v16h b1 = load_frag_b(Bb + 16 * HD,  HD, ncol, grp, k0);
    c[0][0] = WMMA_F16(a0, b0, c[0][0]);
    c[0][1] = WMMA_F16(a0, b1, c[0][1]);
    c[1][0] = WMMA_F16(a1, b0, c[1][0]);
    c[1][1] = WMMA_F16(a1, b1, c[1][1]);
  }
  float* o = out + ((size_t)h * 4096 + (size_t)qh * 64 + qwg * 32) * 64 + khg * 32;
#pragma unroll
  for (int mt = 0; mt < 2; mt++)
#pragma unroll
    for (int nt = 0; nt < 2; nt++)
#pragma unroll
      for (int i = 0; i < 8; i++)
        o[(size_t)(mt * 16 + i + 8 * grp) * 64 + nt * 16 + ncol] = c[mt][nt][i];
}

// --- rel_w[h][qh*64+qw][kw] = sum_d q[h][qh*64+qw][d] * Rw[qw][kw][d] -------
// Fix qw per tile; rows are qh (row stride 4096 halfs in q).
__global__ __launch_bounds__(128) void k_relw_gemm(
    const _Float16* __restrict__ Q, const _Float16* __restrict__ Rw,
    float* __restrict__ out) {
  int wid = threadIdx.x >> 5, lane = threadIdx.x & 31;
  int g = blockIdx.x * 4 + wid;                // 3072 = 12h * 64qw * 4 sub
  int h = g / 256, rem = g % 256;
  int qw = rem / 4, sub = rem % 4;
  int qhg = sub >> 1, kwg = sub & 1;
  int mrow = lane & 15, grp = (lane >> 4) & 1, ncol = lane & 15;

  const _Float16* Ab = Q + (size_t)h * HEAD_E + (size_t)qw * HD + (size_t)qhg * 32 * 4096;
  const _Float16* Bb = Rw + (size_t)qw * 4096 + (size_t)kwg * 32 * HD;
  v8f c[2][2] = {};
#pragma unroll
  for (int k0 = 0; k0 < HD; k0 += 32) {
    v16h a0 = load_frag_a(Ab,              4096, mrow, grp, k0);
    v16h a1 = load_frag_a(Ab + 16 * 4096,  4096, mrow, grp, k0);
    v16h b0 = load_frag_b(Bb,              HD, ncol, grp, k0);
    v16h b1 = load_frag_b(Bb + 16 * HD,    HD, ncol, grp, k0);
    c[0][0] = WMMA_F16(a0, b0, c[0][0]);
    c[0][1] = WMMA_F16(a0, b1, c[0][1]);
    c[1][0] = WMMA_F16(a1, b0, c[1][0]);
    c[1][1] = WMMA_F16(a1, b1, c[1][1]);
  }
  float* o = out + (size_t)h * HEAD_E + (size_t)qhg * 32 * 4096 + (size_t)qw * 64 + kwg * 32;
#pragma unroll
  for (int mt = 0; mt < 2; mt++)
#pragma unroll
    for (int nt = 0; nt < 2; nt++)
#pragma unroll
      for (int i = 0; i < 8; i++)
        o[(size_t)(mt * 16 + i + 8 * grp) * 4096 + nt * 16 + ncol] = c[mt][nt][i];
}

// --- flash attention: one wave = 32 queries of one head ---------------------
// K fragments reused by both q-tiles; V fragments reused by both P-tiles.
__global__ __launch_bounds__(128) void k_attn(
    const _Float16* __restrict__ Q, const _Float16* __restrict__ K,
    const _Float16* __restrict__ VT, const float* __restrict__ RELH,
    const float* __restrict__ RELW, _Float16* __restrict__ O) {
  __shared__ alignas(16) _Float16 pbuf[4][2][16][32];
  int wid = threadIdx.x >> 5, lane = threadIdx.x & 31;
  int g = blockIdx.x * 4 + wid;                // 1536 = 12 heads * 128 q-blocks
  int h = g >> 7, q0 = (g & 127) * 32;
  int mrow = lane & 15, grp = (lane >> 4) & 1, ncol = lane & 15;

  const _Float16* Qh = Q  + (size_t)h * HEAD_E;
  const _Float16* Kh = K  + (size_t)h * HEAD_E;   // pre-scaled by 1/8
  const _Float16* Vh = VT + (size_t)h * HEAD_E;   // [d][s]
  const float*    RH = RELH + (size_t)h * HEAD_E; // [q][kh]
  const float*    RW = RELW + (size_t)h * HEAD_E; // [q][kw]

  v16h aq[4];
  aq[0] = load_frag_a(Qh + (size_t)q0 * HD,        HD, mrow, grp, 0);
  aq[1] = load_frag_a(Qh + (size_t)q0 * HD,        HD, mrow, grp, 32);
  aq[2] = load_frag_a(Qh + (size_t)(q0 + 16) * HD, HD, mrow, grp, 0);
  aq[3] = load_frag_a(Qh + (size_t)(q0 + 16) * HD, HD, mrow, grp, 32);

  float m[2][8], l[2][8];
  v8f acc[2][4] = {};
#pragma unroll
  for (int t = 0; t < 2; t++)
#pragma unroll
    for (int r = 0; r < 8; r++) { m[t][r] = -1e30f; l[t][r] = 0.0f; }

  for (int kb = 0; kb < S_LEN; kb += 32) {
    // shared K fragments for this 32-key block (K dim = head_dim 64)
    v16h bk0 = load_frag_b(Kh + (size_t)kb * HD,        HD, ncol, grp, 0);
    v16h bk1 = load_frag_b(Kh + (size_t)kb * HD,        HD, ncol, grp, 32);
    v16h bk2 = load_frag_b(Kh + (size_t)(kb + 16) * HD, HD, ncol, grp, 0);
    v16h bk3 = load_frag_b(Kh + (size_t)(kb + 16) * HD, HD, ncol, grp, 32);
    if (kb + 32 < S_LEN) __builtin_prefetch(Kh + (size_t)(kb + 32) * HD, 0, 3);

    v8f sc[2][2];
#pragma unroll
    for (int t = 0; t < 2; t++) {
      v8f z0 = {}, z1 = {};
      z0 = WMMA_F16(aq[2 * t],     bk0, z0);
      z0 = WMMA_F16(aq[2 * t + 1], bk1, z0);
      z1 = WMMA_F16(aq[2 * t],     bk2, z1);
      z1 = WMMA_F16(aq[2 * t + 1], bk3, z1);
      sc[t][0] = z0; sc[t][1] = z1;
    }

    int kh = kb >> 6, kw0 = kb & 63;             // 32-key block never crosses a row of 64
#pragma unroll
    for (int t = 0; t < 2; t++) {
#pragma unroll
      for (int r = 0; r < 8; r++) {
        int qrow = q0 + 16 * t + r + 8 * grp;
        float bh = RH[(size_t)qrow * 64 + kh];
        float s0 = sc[t][0][r] + bh + RW[(size_t)qrow * 64 + kw0 + ncol];
        float s1 = sc[t][1][r] + bh + RW[(size_t)qrow * 64 + kw0 + 16 + ncol];
        float mx = fmaxf(s0, s1);
        mx = fmaxf(mx, __shfl_xor(mx, 1));
        mx = fmaxf(mx, __shfl_xor(mx, 2));
        mx = fmaxf(mx, __shfl_xor(mx, 4));
        mx = fmaxf(mx, __shfl_xor(mx, 8));
        float mn = fmaxf(m[t][r], mx);
        float p0 = __expf(s0 - mn), p1 = __expf(s1 - mn);
        float rs = p0 + p1;
        rs += __shfl_xor(rs, 1);
        rs += __shfl_xor(rs, 2);
        rs += __shfl_xor(rs, 4);
        rs += __shfl_xor(rs, 8);
        float al = __expf(m[t][r] - mn);
        l[t][r] = l[t][r] * al + rs;
        m[t][r] = mn;
        acc[t][0][r] *= al; acc[t][1][r] *= al;
        acc[t][2][r] *= al; acc[t][3][r] *= al;
        pbuf[wid][t][r + 8 * grp][ncol]      = (_Float16)p0;
        pbuf[wid][t][r + 8 * grp][16 + ncol] = (_Float16)p1;
      }
    }
    __builtin_amdgcn_wave_barrier();             // order LDS stores before loads (same wave)

    v16h ap0 = load_frag_a(&pbuf[wid][0][0][0], 32, mrow, grp, 0);
    v16h ap1 = load_frag_a(&pbuf[wid][1][0][0], 32, mrow, grp, 0);
#pragma unroll
    for (int nt = 0; nt < 4; nt++) {
      v16h bv = load_frag_b(Vh + (size_t)nt * 16 * S_LEN + kb, S_LEN, ncol, grp, 0);
      acc[0][nt] = WMMA_F16(ap0, bv, acc[0][nt]);
      acc[1][nt] = WMMA_F16(ap1, bv, acc[1][nt]);
    }
  }

#pragma unroll
  for (int t = 0; t < 2; t++) {
#pragma unroll
    for (int r = 0; r < 8; r++) {
      float inv = 1.0f / l[t][r];
      int s = q0 + 16 * t + r + 8 * grp;
#pragma unroll
      for (int nt = 0; nt < 4; nt++)
        O[(size_t)s * CDIM + h * HD + nt * 16 + ncol] = (_Float16)(acc[t][nt][r] * inv);
    }
  }
}

// --- output projection: d_out = AO * Wo^T + b_o (fp32 out), 2x2 tiles ------
__global__ __launch_bounds__(128) void k_out_gemm(
    const _Float16* __restrict__ A, const _Float16* __restrict__ W,
    const float* __restrict__ bias, float* __restrict__ out) {
  int wid = threadIdx.x >> 5, lane = threadIdx.x & 31;
  int g = blockIdx.x * 4 + wid;                // 3072 waves: 128 Mg x 24 Ng
  int m0 = (g / 24) * 32, n0 = (g % 24) * 32;
  int mrow = lane & 15, grp = (lane >> 4) & 1, ncol = lane & 15;

  const _Float16* A0 = A + (size_t)m0 * CDIM;
  const _Float16* A1 = A + (size_t)(m0 + 16) * CDIM;
  const _Float16* B0 = W + (size_t)n0 * CDIM;
  const _Float16* B1 = W + (size_t)(n0 + 16) * CDIM;
  v8f c[2][2] = {};
#pragma unroll 2
  for (int k0 = 0; k0 < CDIM; k0 += 32) {
    v16h a0 = load_frag_a(A0, CDIM, mrow, grp, k0);
    v16h a1 = load_frag_a(A1, CDIM, mrow, grp, k0);
    v16h b0 = load_frag_b(B0, CDIM, ncol, grp, k0);
    v16h b1 = load_frag_b(B1, CDIM, ncol, grp, k0);
    c[0][0] = WMMA_F16(a0, b0, c[0][0]);
    c[0][1] = WMMA_F16(a0, b1, c[0][1]);
    c[1][0] = WMMA_F16(a1, b0, c[1][0]);
    c[1][1] = WMMA_F16(a1, b1, c[1][1]);
  }
#pragma unroll
  for (int nt = 0; nt < 2; nt++) {
    int n = n0 + nt * 16 + ncol;
    float bv = bias[n];
#pragma unroll
    for (int mt = 0; mt < 2; mt++)
#pragma unroll
      for (int i = 0; i < 8; i++)
        out[(size_t)(m0 + mt * 16 + i + 8 * grp) * CDIM + n] = c[mt][nt][i] + bv;
  }
}

// ---------------------------------------------------------------------------
extern "C" void kernel_launch(void* const* d_in, const int* in_sizes, int n_in,
                              void* d_out, int out_size, void* d_ws, size_t ws_size,
                              hipStream_t stream) {
  const float* hid = (const float*)d_in[0];
  const float* w_q = (const float*)d_in[1];  const float* b_q = (const float*)d_in[2];
  const float* w_k = (const float*)d_in[3];  const float* b_k = (const float*)d_in[4];
  const float* w_v = (const float*)d_in[5];  const float* b_v = (const float*)d_in[6];
  const float* w_o = (const float*)d_in[7];  const float* b_o = (const float*)d_in[8];
  const float* rph = (const float*)d_in[9];
  const float* rpw = (const float*)d_in[10];

  // workspace partition (bytes, 256-aligned by construction)
  char* ws = (char*)d_ws;
  size_t off = 0;
  _Float16* Xf  = (_Float16*)(ws + off); off += (size_t)S_LEN * CDIM * 2;
  _Float16* Wq  = (_Float16*)(ws + off); off += (size_t)CDIM * CDIM * 2;
  _Float16* Wk  = (_Float16*)(ws + off); off += (size_t)CDIM * CDIM * 2;
  _Float16* Wv  = (_Float16*)(ws + off); off += (size_t)CDIM * CDIM * 2;
  _Float16* Wo  = (_Float16*)(ws + off); off += (size_t)CDIM * CDIM * 2;
  _Float16* Qf  = (_Float16*)(ws + off); off += (size_t)HEADS * HEAD_E * 2;
  _Float16* Kf  = (_Float16*)(ws + off); off += (size_t)HEADS * HEAD_E * 2;
  _Float16* Vt  = (_Float16*)(ws + off); off += (size_t)HEADS * HEAD_E * 2;
  _Float16* Rhf = (_Float16*)(ws + off); off += (size_t)64 * 64 * 64 * 2;
  _Float16* Rwf = (_Float16*)(ws + off); off += (size_t)64 * 64 * 64 * 2;
  float*    Rlh = (float*)(ws + off);    off += (size_t)HEADS * HEAD_E * 4;
  float*    Rlw = (float*)(ws + off);    off += (size_t)HEADS * HEAD_E * 4;
  _Float16* AO  = (_Float16*)(ws + off); off += (size_t)S_LEN * CDIM * 2;
  (void)ws_size; (void)in_sizes; (void)n_in; (void)out_size;

  const int NX = S_LEN * CDIM;      // 3145728
  const int NW = CDIM * CDIM;       // 589824
  k_f32_to_f16<<<(NX + 255) / 256, 256, 0, stream>>>(hid, Xf, NX);
  k_f32_to_f16<<<(NW + 255) / 256, 256, 0, stream>>>(w_q, Wq, NW);
  k_f32_to_f16<<<(NW + 255) / 256, 256, 0, stream>>>(w_k, Wk, NW);
  k_f32_to_f16<<<(NW + 255) / 256, 256, 0, stream>>>(w_v, Wv, NW);
  k_f32_to_f16<<<(NW + 255) / 256, 256, 0, stream>>>(w_o, Wo, NW);
  k_gather_rel<<<1024, 256, 0, stream>>>(rph, Rhf);
  k_gather_rel<<<1024, 256, 0, stream>>>(rpw, Rwf);

  k_qkv_gemm<<<768, 128, 0, stream>>>(Xf, Wq, b_q, Qf, 0);
  k_qkv_gemm<<<768, 128, 0, stream>>>(Xf, Wk, b_k, Kf, 1);   // scale 1/8 folded in
  k_qkv_gemm<<<768, 128, 0, stream>>>(Xf, Wv, b_v, Vt, 2);   // transposed store

  k_relh_gemm<<<768, 128, 0, stream>>>(Qf, Rhf, Rlh);
  k_relw_gemm<<<768, 128, 0, stream>>>(Qf, Rwf, Rlw);

  k_attn<<<384, 128, 0, stream>>>(Qf, Kf, Vt, Rlh, Rlw, AO);

  k_out_gemm<<<768, 128, 0, stream>>>(AO, Wo, b_o, (float*)d_out);
}
